// InteractionNetwork_66374424592615
// MI455X (gfx1250) — matrix operand order repaired and checked by reading, not compiled, not run on hardware
//
#include <hip/hip_runtime.h>
#include <hip/hip_bf16.h>
#include <math.h>

typedef __attribute__((ext_vector_type(16))) __bf16 v16bf;
typedef __attribute__((ext_vector_type(2)))  __bf16 v2bf;
typedef __attribute__((ext_vector_type(8)))  float  v8f;

#define N_BODY 2048
#define HID 64
#define EPS_SOFT 0.01f
#define BLOCKS_PER_BATCH 256
#define THREADS 256
#define WAVES (THREADS / 32)

union BFrag {
    v16bf v;
    v2bf  p[8];
};

// Packed bf16 ReLU: v_pk_max_num_bf16 (VOP3P op 44). __builtin_elementwise_max
// fails to legalize to the packed op (scalarizes via f32), so emit it directly.
static __device__ __forceinline__ v2bf pk_relu(v2bf x, int zero_reg) {
    int xi = __builtin_bit_cast(int, x);
    int r;
    asm("v_pk_max_num_bf16 %0, %1, %2" : "=v"(r) : "v"(xi), "v"(zero_reg));
    return __builtin_bit_cast(v2bf, r);
}

// Single v_max_num_f32 without the canonicalize max the compiler inserts for
// fmaxf (fmed3 gets const-folded back to maxnum, so asm is the only way).
static __device__ __forceinline__ float relu_f32(float x) {
    float r;
    asm("v_max_num_f32 %0, 0, %1" : "=v"(r) : "v"(x));
    return r;
}

// offset of first edge of row i in triu_indices(N, k=1) order; fits in int32
// for N = 2048 (max value 2,096,128).
static __device__ __forceinline__ int tri_off(int i) {
    return i * (2 * N_BODY - 1 - i) / 2;
}

__global__ void in_zero_kernel(float* __restrict__ out, int n) {
    int idx = blockIdx.x * blockDim.x + threadIdx.x;
    if (idx < n) out[idx] = 0.0f;
}

__global__ __launch_bounds__(THREADS)
void in_edge_kernel(const float* __restrict__ pos, const float* __restrict__ W1,
                    const float* __restrict__ b1,  const float* __restrict__ W2,
                    const float* __restrict__ b2,  const float* __restrict__ W3,
                    const float* __restrict__ b3,  float* __restrict__ out) {
    __shared__ float lds_acc[3 * N_BODY];  // per-block force accumulator (one batch)

    const int tid  = threadIdx.x;
    const int lane = tid & 31;
    const int wave = tid >> 5;
    const int b    = blockIdx.y;
    const int nlo  = lane & 15;   // column selector (edge within tile / A-row)
    const int hsel = lane >> 4;   // 0 = lanes 0-15, 1 = lanes 16-31

    for (int idx = tid; idx < 3 * N_BODY; idx += THREADS) lds_acc[idx] = 0.0f;
    __syncthreads();

    const float b3s = b3[0];

    // Transposed orientation: D[n][edges] = W2^T x h1^T.
    // C layout row M = r + 8*hsel is the hidden feature n = t*16 + r + 8*hsel,
    // column N = nlo is the edge. Per-lane constants for bias + layer-3 weights:
    v8f   biasf[4];          // b2 broadcast in C layout (loop-invariant WMMA C)
    float w3c[4][8];         // W3[n] matching acc[t][r]
#pragma unroll
    for (int t = 0; t < 4; ++t)
#pragma unroll
        for (int r = 0; r < 8; ++r) {
            const int n = t * 16 + r + 8 * hsel;
            biasf[t][r] = b2[n];
            w3c[t][r]   = W3[n];
        }

    // -------- layer-1 constants, pre-packed bf16 pairs in registers --------
    // B-fragment (32x16) K mapping: element j -> K = hsel*8 + (j<8 ? j : 8+j), so chunk c
    // pair i covers consecutive hidden indices k0 = c*32 + hsel*8 + 2i + (i>=4 ? 8 : 0).
    v2bf w1a[2][8], w1b[2][8], b1p[2][8];
#pragma unroll
    for (int c = 0; c < 2; ++c)
#pragma unroll
        for (int i = 0; i < 8; ++i) {
            const int k0 = c * 32 + hsel * 8 + 2 * i + ((i >= 4) ? 8 : 0);
            v2bf a, bb, cc;
            a[0]  = (__bf16)W1[k0];        a[1]  = (__bf16)W1[k0 + 1];
            bb[0] = (__bf16)W1[HID + k0];  bb[1] = (__bf16)W1[HID + k0 + 1];
            cc[0] = (__bf16)b1[k0];        cc[1] = (__bf16)b1[k0 + 1];
            w1a[c][i] = a; w1b[c][i] = bb; b1p[c][i] = cc;
        }

    // Resident W2^T A-fragments (bf16), CDNA5 16x32 16-bit A layout:
    // lane holds row m = nlo (output feature n = t*16 + m);
    // element j -> K = hsel*8 + (j<8 ? j : 8+j); k = c*32 + K.
    v16bf afragW2[2][4];
#pragma unroll
    for (int c = 0; c < 2; ++c)
#pragma unroll
        for (int t = 0; t < 4; ++t) {
#pragma unroll
            for (int j = 0; j < 16; ++j) {
                const int K = hsel * 8 + ((j < 8) ? j : (8 + j));
                const int k = c * 32 + K;
                afragW2[c][t][j] = (__bf16)W2[k * HID + t * 16 + nlo];
            }
        }

    // All edge bookkeeping fits int32 (E = 2,096,128 < 2^31).
    const int E   = N_BODY * (N_BODY - 1) / 2;
    const int epb = (E + gridDim.x - 1) / gridDim.x;
    const int blk_start = (int)blockIdx.x * epb;
    int blk_end = blk_start + epb;
    if (blk_end > E) blk_end = E;

    const float* posb = pos + (size_t)b * N_BODY * 3;
    const int zero_reg = 0;  // packed bf16 {0,0} for ReLU

    for (int tile = blk_start + wave * 16; tile < blk_end; tile += WAVES * 16) {
        // ---- edge features (lanes 0..15 own one edge each) ----
        float d = 0.0f, q = 0.0f;
        float rx = 0.0f, ry = 0.0f, rz = 0.0f;
        int ii = 0, jj = 0;
        bool evalid = false;
        const int e = tile + lane;
        if (lane < 16 && e < blk_end) {
            evalid = true;
            // float inversion is exact enough: (2N-1)^2 = 16,769,025 < 2^24,
            // 8e <= 16,769,024 < 2^24, and the while-loops fix any +-1 error.
            const float tt   = 2.0f * N_BODY - 1.0f;
            const float disc = tt * tt - 8.0f * (float)e;
            int i = (int)((tt - sqrtf(disc)) * 0.5f);
            if (i < 0) i = 0;
            if (i > N_BODY - 2) i = N_BODY - 2;
            while (tri_off(i + 1) <= e) ++i;
            while (tri_off(i) > e) --i;
            const int j = e - tri_off(i) + i + 1;
            ii = i; jj = j;
            rx = posb[3 * j + 0] - posb[3 * i + 0];
            ry = posb[3 * j + 1] - posb[3 * i + 1];
            rz = posb[3 * j + 2] - posb[3 * i + 2];
            const float d2 = rx * rx + ry * ry + rz * rz + EPS_SOFT * EPS_SOFT;
            d = sqrtf(d2);
            q = 1.0f / (d2 * d);
        }
        // every lane fetches the features of the edge column it holds (e = nlo)
        const float dm = __shfl(d, nlo, 32);
        const float qm = __shfl(q, nlo, 32);

        // ---- layer 1: packed bf16 fma + packed ReLU into B-fragment layout ----
        v2bf dm2, qm2;
        dm2[0] = (__bf16)dm; dm2[1] = (__bf16)dm;
        qm2[0] = (__bf16)qm; qm2[1] = (__bf16)qm;

        BFrag hb[2];
#pragma unroll
        for (int c = 0; c < 2; ++c) {
#pragma unroll
            for (int i = 0; i < 8; ++i) {
                v2bf t = __builtin_elementwise_fma(qm2, w1b[c][i], b1p[c][i]);
                t = __builtin_elementwise_fma(dm2, w1a[c][i], t);
                hb[c].p[i] = pk_relu(t, zero_reg);
            }
        }

        // ---- layer 2 (transposed): 4 n-tiles x 2 K-chunks of WMMA ----
        v8f acc[4];
#pragma unroll
        for (int t = 0; t < 4; ++t) {
            acc[t] = __builtin_amdgcn_wmma_f32_16x16x32_bf16(
                false, afragW2[0][t], false, hb[0].v, (short)0, biasf[t], false, false);
        }
#pragma unroll
        for (int t = 0; t < 4; ++t) {
            acc[t] = __builtin_amdgcn_wmma_f32_16x16x32_bf16(
                false, afragW2[1][t], false, hb[1].v, (short)0, acc[t], false, false);
        }

        // ---- relu + layer-3 dot: 4 independent chains + tree combine ----
        float sp[4];
#pragma unroll
        for (int t = 0; t < 4; ++t) {
            float st = 0.0f;
#pragma unroll
            for (int r = 0; r < 8; ++r)
                st = fmaf(relu_f32(acc[t][r]), w3c[t][r], st);
            sp[t] = st;
        }
        float s = (sp[0] + sp[1]) + (sp[2] + sp[3]);

        // merge the two half-wave feature halves of the same edge column
        s += __shfl_xor(s, 16, 32);
        const float mval = s + b3s;  // lanes 0..15 now hold m_ij for their edge

        // ---- forces: Newton's third law scatter into LDS (ds_add_f32) ----
        if (evalid) {
            const float fx = mval * rx, fy = mval * ry, fz = mval * rz;
            atomicAdd(&lds_acc[3 * ii + 0], fx);
            atomicAdd(&lds_acc[3 * ii + 1], fy);
            atomicAdd(&lds_acc[3 * ii + 2], fz);
            atomicAdd(&lds_acc[3 * jj + 0], -fx);
            atomicAdd(&lds_acc[3 * jj + 1], -fy);
            atomicAdd(&lds_acc[3 * jj + 2], -fz);
        }
    }

    __syncthreads();
    float* outb = out + (size_t)b * N_BODY * 3;
    for (int idx = tid; idx < 3 * N_BODY; idx += THREADS)
        atomicAdd(&outb[idx], lds_acc[idx]);
}

extern "C" void kernel_launch(void* const* d_in, const int* in_sizes, int n_in,
                              void* d_out, int out_size, void* d_ws, size_t ws_size,
                              hipStream_t stream) {
    const float* pos = (const float*)d_in[0];
    const float* W1  = (const float*)d_in[1];
    const float* b1  = (const float*)d_in[2];
    const float* W2  = (const float*)d_in[3];
    const float* b2  = (const float*)d_in[4];
    const float* W3  = (const float*)d_in[5];
    const float* b3  = (const float*)d_in[6];
    float* out = (float*)d_out;

    const int B = in_sizes[0] / (N_BODY * 3);  // 2

    in_zero_kernel<<<(out_size + 255) / 256, 256, 0, stream>>>(out, out_size);

    dim3 grid(BLOCKS_PER_BATCH, B);
    in_edge_kernel<<<grid, THREADS, 0, stream>>>(pos, W1, b1, W2, b2, W3, b3, out);
}
